// OuterProductUpdate_88948772700513
// MI455X (gfx1250) — matrix operand order repaired and checked by reading, not compiled, not run on hardware
//
#include <hip/hip_runtime.h>

#define NN    384
#define DNODE 256
#define DPAIR 128
#define DHID  32
#define EPS   1e-5f

typedef __attribute__((ext_vector_type(16))) _Float16 v16h;
typedef __attribute__((ext_vector_type(8)))  _Float16 v8h;
typedef __attribute__((ext_vector_type(8)))  float    v8f;

// ---------------------------------------------------------------------------
// Stage 1: LayerNorm + projections a = sn*Wa^T + ba (f32), b = sn*Wb^T + bb
//          (stored directly as f16 in WMMA-friendly row-major [N][32]).
// One block per row i, 256 threads.
// ---------------------------------------------------------------------------
__global__ __launch_bounds__(256)
void ln_proj_kernel(const float* __restrict__ s,
                    const float* __restrict__ gamma,
                    const float* __restrict__ beta,
                    const float* __restrict__ Wa, const float* __restrict__ ba,
                    const float* __restrict__ Wb, const float* __restrict__ bb,
                    float*    __restrict__ a_out,   // [N][DHID] f32
                    _Float16* __restrict__ b_half)  // [N][DHID] f16
{
    __shared__ float red[DNODE];
    __shared__ float snL[DNODE];
    const int i = blockIdx.x;
    const int t = threadIdx.x;

    const float x = s[i * DNODE + t];
    red[t] = x;
    __syncthreads();
    for (int off = 128; off > 0; off >>= 1) {
        if (t < off) red[t] += red[t + off];
        __syncthreads();
    }
    const float mu = red[0] * (1.0f / DNODE);
    __syncthreads();

    const float d = x - mu;
    red[t] = d * d;
    __syncthreads();
    for (int off = 128; off > 0; off >>= 1) {
        if (t < off) red[t] += red[t + off];
        __syncthreads();
    }
    const float var = red[0] * (1.0f / DNODE);

    snL[t] = d * rsqrtf(var + EPS) * gamma[t] + beta[t];
    __syncthreads();

    if (t < 64) {
        const int h = t & 31;
        const float* __restrict__ W = (t < 32) ? Wa : Wb;
        float sum = 0.0f;
        #pragma unroll 8
        for (int dd = 0; dd < DNODE; ++dd) sum += snL[dd] * W[h * DNODE + dd];
        if (t < 32) a_out[i * DHID + h]  = sum + ba[h];
        else        b_half[i * DHID + h] = (_Float16)(sum + bb[h]);
    }
}

// ---------------------------------------------------------------------------
// Stage 2: T[i,p,e] = sum_d a[i,d] * Wo[p, d*32 + e], stored f16 row-major
//          [N][DPAIR][DHID].  One block per i; Wo (512KB) stays L2-resident.
// ---------------------------------------------------------------------------
__global__ __launch_bounds__(256)
void make_T_kernel(const float* __restrict__ a,   // [N][DHID]
                   const float* __restrict__ Wo,  // [DPAIR][DHID*DHID]
                   _Float16*    __restrict__ T)   // [N][DPAIR][DHID]
{
    __shared__ float aL[DHID];
    const int i = blockIdx.x;
    const int t = threadIdx.x;
    if (t < DHID) aL[t] = a[i * DHID + t];
    __syncthreads();

    for (int k = t; k < DPAIR * DHID; k += 256) {
        const int p = k >> 5;
        const int e = k & 31;
        const float* __restrict__ w = Wo + p * (DHID * DHID) + e;
        float sum = 0.0f;
        #pragma unroll
        for (int dd = 0; dd < DHID; ++dd) sum += aL[dd] * w[dd * DHID];
        T[(i * DPAIR + p) * DHID + e] = (_Float16)sum;
    }
}

// ---------------------------------------------------------------------------
// Stage 3: out[i, j, p] = sum_e b[j,e] * T[i,p,e] + bo[p]
// One v_wmma_f32_16x16x32_f16 per 16(j) x 16(p) output tile (K = 32 = DHID).
// Grid: (i, j-slab of 64 rows); 8 waves per block: wave&3 -> j-tile,
// wave>>2 selects base of 4 fully-unrolled p-tiles (stride 2 tiles).  All
// B-tile loads, bias loads and output stores are expressed relative to one
// base pointer per stream so they lower to immediate-offset global ops
// (offsets q*2048B / q*128B / q*128+v*512B all fit the 24-bit IOFFSET).
// Bias bo[p] is seeded into the C accumulator so WMMA performs the add.
// WMMA D layout: lane%16 = p col (contiguous 64B store segments per
// half-wave), VGPR v + 8*(lane/16) = j row.
// ---------------------------------------------------------------------------
__global__ __launch_bounds__(256)
void gemm_wmma_kernel(const _Float16* __restrict__ bh, // [N][DHID]
                      const _Float16* __restrict__ T,  // [N][DPAIR][DHID]
                      const float*    __restrict__ bo, // [DPAIR]
                      float*          __restrict__ out) // [N][N][DPAIR]
{
    const int i      = blockIdx.x;
    const int wave   = threadIdx.x >> 5;
    const int lane   = threadIdx.x & 31;
    const int l16    = lane & 15;
    const int half16 = lane >> 4;          // 0 or 1
    const int j0     = (blockIdx.y * 4 + (wave & 3)) * 16;
    const int ptbase = wave >> 2;          // 0 or 1; p-tiles ptbase, +2, +4, +6

    // A operand (16x32 f16, contiguous k order, same as B): lane holds row M = l16;
    // K = half16*16 .. +15 contiguous in VGPRs 0-7.
    const _Float16* __restrict__ abase = bh + (j0 + l16) * DHID + half16 * 16;
    const v8h alo = *(const v8h*)(abase);
    const v8h ahi = *(const v8h*)(abase + 8);
    v16h A;
    #pragma unroll
    for (int k = 0; k < 8; ++k) { A[k] = alo[k]; A[k + 8] = ahi[k]; }

    // Single base pointers; all per-q accesses are constant immediate offsets.
    const _Float16* __restrict__ tbase =
        T + ((size_t)(i * DPAIR + ptbase * 16 + l16)) * DHID + half16 * 16;
    const float* __restrict__ bbase = bo + ptbase * 16 + l16;
    float* __restrict__ obase =
        out + ((size_t)i * NN + j0 + half16 * 8) * DPAIR + ptbase * 16 + l16;

    // Prefetch all 4 B operands and biases up front (fully unrolled: uniform
    // control flow, EXEC stays all-1s for every WMMA).
    v16h  Bt[4];
    float bias[4];
    #pragma unroll
    for (int q = 0; q < 4; ++q) {
        // B operand (32x16 f16, contiguous k order, same as A): lane holds col N = p0+l16;
        // K = half16*16 .. +15 contiguous in VGPRs 0-7.
        Bt[q]   = *(const v16h*)(tbase + q * (2 * 16 * DHID)); // +2048 B per q
        bias[q] = bbase[q * 32];                               // +128 B per q
    }

    #pragma unroll
    for (int q = 0; q < 4; ++q) {
        v8f c;
        #pragma unroll
        for (int v = 0; v < 8; ++v) c[v] = bias[q];   // seed accumulator with bo[p]
        c = __builtin_amdgcn_wmma_f32_16x16x32_f16(
                /*neg_a=*/false, A, /*neg_b=*/false, Bt[q],
                /*c_mod=*/(short)0, c, /*reuse_a=*/false, /*reuse_b=*/false);

        #pragma unroll
        for (int v = 0; v < 8; ++v)
            obase[q * 32 + v * DPAIR] = c[v];          // +128*q + 512*v bytes
    }
}

// ---------------------------------------------------------------------------
extern "C" void kernel_launch(void* const* d_in, const int* in_sizes, int n_in,
                              void* d_out, int out_size, void* d_ws, size_t ws_size,
                              hipStream_t stream) {
    const float* s     = (const float*)d_in[0];
    const float* gamma = (const float*)d_in[1];
    const float* beta  = (const float*)d_in[2];
    const float* Wa    = (const float*)d_in[3];
    const float* ba    = (const float*)d_in[4];
    const float* Wb    = (const float*)d_in[5];
    const float* bb    = (const float*)d_in[6];
    const float* Wo    = (const float*)d_in[7];
    const float* bo    = (const float*)d_in[8];
    float* out = (float*)d_out;

    char* ws = (char*)d_ws;
    float*    a_f32 = (float*)ws;                          // 384*32*4      = 49152 B
    _Float16* T     = (_Float16*)(ws + 49152);             // 384*128*32*2  = 3145728 B
    _Float16* bh    = (_Float16*)(ws + 49152 + 3145728);   // 384*32*2      = 24576 B

    ln_proj_kernel<<<NN, 256, 0, stream>>>(s, gamma, beta, Wa, ba, Wb, bb, a_f32, bh);
    make_T_kernel<<<NN, 256, 0, stream>>>(a_f32, Wo, T);
    gemm_wmma_kernel<<<dim3(NN, 6), 256, 0, stream>>>(bh, T, bo, out);
}
